// LocalCSAttention_70463233458348
// MI455X (gfx1250) — matrix operand order repaired and checked by reading, not compiled, not run on hardware
//
#include <hip/hip_runtime.h>

// LocalCSAttention for MI455X (gfx1250, wave32, WMMA + TDM async tensor loads).
// ws layout assumption: ~880 MB scratch (compile-only environment).

#define DEV __device__ __forceinline__

typedef __bf16 bf16_t;
typedef bf16_t v16bf __attribute__((ext_vector_type(16)));
typedef bf16_t bf16x8 __attribute__((ext_vector_type(8)));
typedef float  v8f   __attribute__((ext_vector_type(8)));
typedef float  v2f   __attribute__((ext_vector_type(2)));
typedef unsigned int u32x4 __attribute__((ext_vector_type(4)));
typedef int i32x4 __attribute__((ext_vector_type(4)));
typedef int i32x8 __attribute__((ext_vector_type(8)));

constexpr int    IMG  = 256;
constexpr size_t PIX  = 65536;   // 256*256

DEV v8f wmma_bf16(v16bf a, v16bf b, v8f c) {
  return __builtin_amdgcn_wmma_f32_16x16x32_bf16(false, a, false, b, (short)0, c, false, false);
}
DEV v8f wmma_f32(v2f a, v2f b, v8f c) {
  return __builtin_amdgcn_wmma_f32_16x16x4_f32(false, a, false, b, (short)0, c, false, false);
}
DEV int regc(int i) { return (i < 248) ? 0 : ((i < 252) ? 1 : 2); }

// ---------------------------------------------------------------------------
// Tensor Data Mover: async 2D fp32 tile load (global -> LDS).
// D# per CDNA5 ISA ch.8: group0 {count, lds_addr, global_addr, type=2},
// group1 {data_size=4B, tensor dims, tile dims, dim0 stride}.
// 6-arg builtin form (amdgpu-toolchain / clang-23).
// ---------------------------------------------------------------------------
DEV void tdm_load_tile_f32(const float* gsrc, unsigned lds_byte_off,
                           unsigned tile_x, unsigned tile_y,
                           unsigned tensor_x, unsigned tensor_y,
                           unsigned long long stride_elems) {
  unsigned long long ga = (unsigned long long)(uintptr_t)gsrc;
  u32x4 g0 = {};
  g0[0] = 1u;                                                 // count=1 (valid D#)
  g0[1] = lds_byte_off;                                       // lds_addr
  g0[2] = (unsigned)ga;                                       // global_addr[31:0]
  g0[3] = (unsigned)((ga >> 32) & 0x01FFFFFFu) | (2u << 30);  // addr[56:32] | type=2
  i32x8 g1 = {};
  g1[0] = (int)(2u << 16);                                    // data_size = 4 bytes
  g1[1] = (int)((tensor_x & 0xFFFFu) << 16);                  // tensor_dim0[15:0]
  g1[2] = (int)(((tensor_x >> 16) & 0xFFFFu) |                // tensor_dim0[31:16]
                ((tensor_y & 0xFFFFu) << 16));                // tensor_dim1[15:0]
  g1[3] = (int)(((tensor_y >> 16) & 0xFFFFu) |                // tensor_dim1[31:16]
                ((tile_x & 0xFFFFu) << 16));                  // tile_dim0
  g1[4] = (int)(tile_y & 0xFFFFu);                            // tile_dim1 (tile_dim2=0)
  g1[5] = (int)(unsigned)(stride_elems & 0xFFFFFFFFu);        // tensor_dim0_stride lo
  g1[6] = (int)(unsigned)((stride_elems >> 32) & 0xFFFFu);    // tensor_dim0_stride hi
  i32x4 gz4 = {};
  i32x8 gz8 = {};
  __builtin_amdgcn_tensor_load_to_lds(g0, g1, gz4, gz4, gz8, 0);
}

// ---------------------------------------------------------------------------
// Generic GEMM  Y[M][N] = W[M][K] * X[K][N]   (bf16 WMMA, fp32 accum)
// block: 256 thr = 8 waves; block tile M=16, N=128; K step 32.
// B tiles arrive via TDM (double-buffered, TENSORcnt-pipelined).
// ---------------------------------------------------------------------------
__global__ __launch_bounds__(256)
void k_gemm_bf16(const float* __restrict__ Wm, const float* __restrict__ X,
                 float* __restrict__ Y, int M, int K, int N) {
  __shared__ __align__(16) bf16_t aS[16 * 32];       // [row][k] weights (bf16)
  __shared__ __align__(16) float  bS[2][32 * 128];   // [k][col] activations (fp32, TDM)
  const int tid  = threadIdx.x;
  const int wave = tid >> 5, lane = tid & 31;
  const int m0 = blockIdx.y * 16;
  const int n0 = blockIdx.x * 128;
  const int row = lane & 15;
  const int half = lane >> 4;
  const int kbA = half * 8;    // A frag: lanes 0-15 hold K 0-7/16-23, lanes 16-31 8-15/24-31
  const int kbB = half * 16;   // B frag: same K order as A frag (lanes 0-15 K 0-7/16-23, 16-31 8-15/24-31)
  const int col = wave * 16 + row;
  const unsigned ldsb[2] = { (unsigned)(uintptr_t)&bS[0][0],
                             (unsigned)(uintptr_t)&bS[1][0] };
  v8f acc = {};
  if (wave == 0)
    tdm_load_tile_f32(X + n0, ldsb[0], 128, 32,
                      (unsigned)(N - n0), (unsigned)K, (unsigned long long)N);
  int cur = 0;
  for (int k0 = 0; k0 < K; k0 += 32) {
    if (wave == 0 && (k0 + 32) < K)
      tdm_load_tile_f32(X + (size_t)(k0 + 32) * N + n0, ldsb[cur ^ 1], 128, 32,
                        (unsigned)(N - n0), (unsigned)(K - k0 - 32),
                        (unsigned long long)N);
    // stage A (16x32 weights, fp32 -> bf16), exactly 2 elements per thread
    {
      int i0 = tid, i1 = tid + 256;
      aS[i0] = (bf16_t)Wm[(size_t)(m0 + (i0 >> 5)) * K + (k0 + (i0 & 31))];
      aS[i1] = (bf16_t)Wm[(size_t)(m0 + (i1 >> 5)) * K + (k0 + (i1 & 31))];
    }
    if (wave == 0) {
      if ((k0 + 32) < K) __builtin_amdgcn_s_wait_tensorcnt(1);  // current tile resident
      else               __builtin_amdgcn_s_wait_tensorcnt(0);
    }
    __syncthreads();
    v16bf a, b;
    {
      bf16x8 lo = *(const bf16x8*)(aS + row * 32 + kbA);
      bf16x8 hi = *(const bf16x8*)(aS + row * 32 + 16 + kbA);
      #pragma unroll
      for (int i = 0; i < 8; ++i) { a[i] = lo[i]; a[8 + i] = hi[i]; }
    }
    {
      const float* bp = &bS[cur][0];
      #pragma unroll
      for (int i = 0; i < 16; ++i)
        b[i] = (bf16_t)bp[(kbA + (i & 7) + ((i >> 3) << 4)) * 128 + col];
    }
    acc = wmma_bf16(a, b, acc);
    __syncthreads();
    cur ^= 1;
  }
  // C/D layout: VGPR v -> M = v (lanes 0-15) or v+8 (lanes 16-31); N = lane&15
  const int oc = n0 + wave * 16 + row;
  #pragma unroll
  for (int v = 0; v < 8; ++v)
    Y[(size_t)(m0 + v + half * 8) * N + oc] = acc[v];
}

// ---------------------------------------------------------------------------
// Windowing: cyclic shift (-4,-4) + to_windows -> Q/K/V [h][n][p][c32]
// ---------------------------------------------------------------------------
__global__ __launch_bounds__(256)
void k_window_qkv(const float* __restrict__ Yq, const float* __restrict__ Ykv,
                  float* __restrict__ Q, float* __restrict__ Kw, float* __restrict__ V) {
  int idx = blockIdx.x * 256 + threadIdx.x;            // (h,n,p)
  int p = idx & 63, n = (idx >> 6) & 1023, h = idx >> 16;
  int wi = n >> 5, wj = n & 31, pi = p >> 3, pj = p & 7;
  int sr = (wi * 8 + pi + 4) & 255, sc = (wj * 8 + pj + 4) & 255;
  size_t src = (size_t)sr * IMG + sc;
  size_t dst = (size_t)idx * 32;
  #pragma unroll
  for (int c = 0; c < 32; ++c) {
    int ch = h * 32 + c;
    Q[dst + c]  = Yq[(size_t)ch * PIX + src];
    Kw[dst + c] = Ykv[(size_t)ch * PIX + src];
    V[dst + c]  = Ykv[(size_t)(256 + ch) * PIX + src];
  }
}

// branch b: norm over c (per h,n,p); branch l: norm over p (per h,n,c)
__global__ void k_norm_bl(float* __restrict__ Q, float* __restrict__ Kw) {
  int hn = blockIdx.x;            // 8192
  int t  = threadIdx.x;           // 64
  size_t base = (size_t)hn * 2048;
  for (int w = 0; w < 2; ++w) {
    float* A = w ? Kw : Q;
    float s = 0.f;
    #pragma unroll
    for (int c = 0; c < 8; ++c) { float v = A[base + t * 32 + c]; s += v * v; }
    float r = 1.0f / fmaxf(sqrtf(s), 1e-12f);
    #pragma unroll
    for (int c = 0; c < 8; ++c) A[base + t * 32 + c] *= r;
  }
  if (t < 16) {
    float* A = (t < 8) ? Q : Kw;
    int c = 8 + (t & 7);
    float s = 0.f;
    for (int p = 0; p < 64; ++p) { float v = A[base + p * 32 + c]; s += v * v; }
    float r = 1.0f / fmaxf(sqrtf(s), 1e-12f);
    for (int p = 0; p < 64; ++p) A[base + p * 32 + c] *= r;
  }
}

// branches g,r: norm across the 1024 windows per (h,p,c). Two passes.
__global__ __launch_bounds__(256)
void k_norm_gr_reduce(const float* __restrict__ Q, const float* __restrict__ Kw,
                      float* __restrict__ nrm) {
  __shared__ float red[256];
  int hp = blockIdx.x;                  // (h,p) 512
  int h = hp >> 6, p = hp & 63;
  int t = threadIdx.x;
  int combo = t >> 3, sub = t & 7;      // combo: which(2b) x c(3b)
  int which = combo >> 3, c = combo & 7;
  const float* A = (which < 2) ? Q : Kw;
  int ch = ((which & 1) ? 24 : 16) + c;
  float s = 0.f;
  for (int n = sub; n < 1024; n += 8) {
    float v = A[((size_t)(h * 1024 + n) * 64 + p) * 32 + ch];
    s += v * v;
  }
  red[t] = s;
  __syncthreads();
  if (sub == 0) {
    #pragma unroll
    for (int i = 1; i < 8; ++i) s += red[t + i];
    nrm[((size_t)which * 8 + h) * 512 + p * 8 + c] = 1.0f / fmaxf(sqrtf(s), 1e-12f);
  }
}

__global__ __launch_bounds__(256)
void k_norm_gr_apply(float* __restrict__ Q, float* __restrict__ Kw,
                     const float* __restrict__ nrm) {
  size_t idx = (size_t)blockIdx.x * 256 + threadIdx.x;   // (h,n,p,c16)
  int c16 = idx & 15; size_t hnp = idx >> 4;
  int p = hnp & 63; size_t hn = hnp >> 6;
  int h = (int)(hn >> 10);
  int wh = c16 >> 3, c = c16 & 7;
  size_t e = hnp * 32 + 16 + c16;
  Q[e]  *= nrm[((size_t)wh * 8 + h) * 512 + p * 8 + c];
  Kw[e] *= nrm[((size_t)(2 + wh) * 8 + h) * 512 + p * 8 + c];
}

// ---------------------------------------------------------------------------
// attn_b scores: per window 64x64 = qb (64x8) . kb^T, f32 WMMA 16x16x4
// ---------------------------------------------------------------------------
__global__ __launch_bounds__(256)
void k_attn_b_score(const float* __restrict__ Q, const float* __restrict__ Kw,
                    const float* __restrict__ temp, float* __restrict__ Sb) {
  int wave = threadIdx.x >> 5, lane = threadIdx.x & 31;
  int hn = blockIdx.x * 8 + wave;       // 8192 windows
  int h = hn >> 10;
  float t0 = temp[h];
  size_t qbase = (size_t)hn * 2048;
  size_t sbase = (size_t)hn * 4096;
  int row = lane & 15, half = lane >> 4;
  for (int mt = 0; mt < 4; ++mt)
    for (int nt = 0; nt < 4; ++nt) {
      v8f acc = {};
      #pragma unroll
      for (int ks = 0; ks < 2; ++ks) {
        int kk = ks * 4 + half * 2;     // 32-bit A 16x4: lanes 0-15 K0,1; 16-31 K2,3
        v2f a, b;
        const float* qp = &Q[qbase + (size_t)(mt * 16 + row) * 32 + kk];
        a[0] = qp[0]; a[1] = qp[1];
        const float* kp = &Kw[qbase + (size_t)(nt * 16 + row) * 32 + kk];
        b[0] = kp[0]; b[1] = kp[1];     // B[k][q] = kb[q][k]
        acc = wmma_f32(a, b, acc);
      }
      #pragma unroll
      for (int v = 0; v < 8; ++v)
        Sb[sbase + (size_t)(mt * 16 + v + half * 8) * 64 + nt * 16 + row] = acc[v] * t0;
    }
}

// talk_conv (3x3 over window grid, head mixing) + mask + softmax over q
__global__ __launch_bounds__(256)
void k_talk_mask_softmax_b(const float* __restrict__ Sb, const float* __restrict__ btw,
                           float* __restrict__ Ab) {
  __shared__ float red[256];
  int t = threadIdx.x;
  int grp = t >> 6, q = t & 63;
  size_t rowid = (size_t)blockIdx.x * 4 + grp;   // (h2,n,p)
  int p = rowid & 63;
  size_t hn = rowid >> 6;
  int n = hn & 1023, h2 = (int)(hn >> 10);
  int wi = n >> 5, wj = n & 31;
  float acc = 0.f;
  for (int h = 0; h < 8; ++h)
    for (int di = -1; di <= 1; ++di) {
      int ni = wi + di; if (ni < 0 || ni >= 32) continue;
      for (int dj = -1; dj <= 1; ++dj) {
        int nj = wj + dj; if (nj < 0 || nj >= 32) continue;
        float w = btw[((h2 * 8 + h) * 3 + (di + 1)) * 3 + (dj + 1)];
        acc += w * Sb[((size_t)(h * 1024 + ni * 32 + nj) * 64 + p) * 64 + q];
      }
    }
  // shifted-window mask on unshifted window coords
  int cp = 3 * regc(wi * 8 + (p >> 3)) + regc(wj * 8 + (p & 7));
  int cq = 3 * regc(wi * 8 + (q >> 3)) + regc(wj * 8 + (q & 7));
  if (cp != cq) acc -= 100.0f;
  red[t] = acc;
  __syncthreads();
  float m = -1e30f;
  for (int i = 0; i < 64; ++i) m = fmaxf(m, red[(grp << 6) + i]);
  float e = __expf(acc - m);
  __syncthreads();
  red[t] = e;
  __syncthreads();
  float sum = 0.f;
  for (int i = 0; i < 64; ++i) sum += red[(grp << 6) + i];
  Ab[rowid * 64 + q] = e / sum;
}

// out_b = Ab(64x64) . vb(64x8), f32 WMMA, fused from_windows + roll store
__global__ __launch_bounds__(256)
void k_out_b(const float* __restrict__ Ab, const float* __restrict__ V,
             float* __restrict__ OutIm) {
  int wave = threadIdx.x >> 5, lane = threadIdx.x & 31;
  int hn = blockIdx.x * 8 + wave;
  int h = hn >> 10, n = hn & 1023;
  int wi = n >> 5, wj = n & 31;
  size_t abase = (size_t)hn * 4096;
  size_t vbase = (size_t)hn * 2048;
  int row = lane & 15, half = lane >> 4;
  for (int mt = 0; mt < 4; ++mt) {
    v8f acc = {};
    #pragma unroll
    for (int ks = 0; ks < 16; ++ks) {
      int kk = ks * 4 + half * 2;
      v2f a, b;
      a[0] = Ab[abase + (size_t)(mt * 16 + row) * 64 + kk];
      a[1] = Ab[abase + (size_t)(mt * 16 + row) * 64 + kk + 1];
      b[0] = (row < 8) ? V[vbase + (size_t)kk * 32 + row] : 0.f;        // B[k][c]=vb[k][c]
      b[1] = (row < 8) ? V[vbase + (size_t)(kk + 1) * 32 + row] : 0.f;
      acc = wmma_f32(a, b, acc);
    }
    if (row < 8) {
      #pragma unroll
      for (int v = 0; v < 8; ++v) {
        int p = mt * 16 + v + half * 8;
        int ir = (wi * 8 + (p >> 3) + 4) & 255, ic = (wj * 8 + (p & 7) + 4) & 255;
        OutIm[(size_t)(h * 8 + row) * PIX + (size_t)ir * IMG + ic] = acc[v];
      }
    }
  }
}

// ---------------------------------------------------------------------------
// small branches l, g, r (VALU; tiny matmuls)
// ---------------------------------------------------------------------------
__global__ void k_attn_l(const float* __restrict__ Q, const float* __restrict__ Kw,
                         const float* __restrict__ temp, float* __restrict__ Sl) {
  int hn = blockIdx.x;
  int t = threadIdx.x;                  // (c,d)
  int c = t >> 3, d = t & 7;
  float t1 = temp[8 + (hn >> 10)];
  size_t base = (size_t)hn * 2048;
  float s = 0.f;
  for (int p = 0; p < 64; ++p)
    s += Q[base + p * 32 + 8 + c] * Kw[base + p * 32 + 8 + d];
  Sl[(size_t)hn * 64 + t] = s * t1;
}

__global__ __launch_bounds__(256)
void k_talk_softmax_l(const float* __restrict__ Sl, const float* __restrict__ ltw,
                      float* __restrict__ Al) {
  int idx = blockIdx.x * 256 + threadIdx.x;     // (h2,n,c) 65536
  int c = idx & 7, n = (idx >> 3) & 1023, h2 = idx >> 13;
  int wi = n >> 5, wj = n & 31;
  float acc[8] = {};
  for (int h = 0; h < 8; ++h)
    for (int di = -1; di <= 1; ++di) {
      int ni = wi + di; if (ni < 0 || ni >= 32) continue;
      for (int dj = -1; dj <= 1; ++dj) {
        int nj = wj + dj; if (nj < 0 || nj >= 32) continue;
        float w = ltw[((h2 * 8 + h) * 3 + (di + 1)) * 3 + (dj + 1)];
        const float* s = &Sl[(size_t)(h * 1024 + ni * 32 + nj) * 64 + c * 8];
        #pragma unroll
        for (int d = 0; d < 8; ++d) acc[d] += w * s[d];
      }
    }
  float m = acc[0];
  #pragma unroll
  for (int d = 1; d < 8; ++d) m = fmaxf(m, acc[d]);
  float sum = 0.f;
  #pragma unroll
  for (int d = 0; d < 8; ++d) { acc[d] = __expf(acc[d] - m); sum += acc[d]; }
  float r = 1.0f / sum;
  float* o = &Al[(size_t)idx * 8];
  #pragma unroll
  for (int d = 0; d < 8; ++d) o[d] = acc[d] * r;
}

__global__ void k_attn_g(const float* __restrict__ Q, const float* __restrict__ Kw,
                         const float* __restrict__ temp, float* __restrict__ Sg) {
  int hp = blockIdx.x;                  // (h,p) 512
  int h = hp >> 6, p = hp & 63;
  float t2 = temp[16 + h];
  int t = threadIdx.x;                  // (c,d)
  int c = t >> 3, d = t & 7;
  float s = 0.f;
  for (int n = 0; n < 1024; ++n) {
    size_t base = ((size_t)(h * 1024 + n) * 64 + p) * 32;
    s += Q[base + 16 + c] * Kw[base + 16 + d];
  }
  Sg[(size_t)hp * 64 + t] = s * t2;     // [h][p][c][d]
}

__global__ void k_attn_r(const float* __restrict__ Q, const float* __restrict__ Kw,
                         const float* __restrict__ temp, float* __restrict__ Sr) {
  int b = blockIdx.x;                   // (h,c,p) 4096
  int p = b & 63, c = (b >> 6) & 7, h = b >> 9;
  float t3 = temp[24 + h];
  int q = threadIdx.x;                  // 64
  float s = 0.f;
  for (int n = 0; n < 1024; ++n) {
    size_t base = (size_t)(h * 1024 + n) * 2048;
    s += Q[base + p * 32 + 24 + c] * Kw[base + q * 32 + 24 + c];
  }
  Sr[(size_t)b * 64 + q] = s * t3;      // [h][c][p][q]
}

__global__ __launch_bounds__(256)
void k_g_talk_softmax(const float* __restrict__ Sg, const float* __restrict__ gt,
                      float* __restrict__ Ag) {
  int idx = blockIdx.x * 256 + threadIdx.x;     // (t,l,c) 4096
  int c = idx & 7, l = (idx >> 3) & 63, tt = idx >> 9;
  float acc[8] = {};
  for (int h = 0; h < 8; ++h)
    for (int k = 0; k < 64; ++k) {
      float w = gt[(((size_t)h * 64 + k) * 64 + l) * 8 + tt];
      const float* s = &Sg[(size_t)(h * 64 + k) * 64 + c * 8];
      #pragma unroll
      for (int d = 0; d < 8; ++d) acc[d] += w * s[d];
    }
  float m = acc[0];
  #pragma unroll
  for (int d = 1; d < 8; ++d) m = fmaxf(m, acc[d]);
  float sum = 0.f;
  #pragma unroll
  for (int d = 0; d < 8; ++d) { acc[d] = __expf(acc[d] - m); sum += acc[d]; }
  float r = 1.0f / sum;
  float* o = &Ag[(size_t)idx * 8];      // [t][l][c][d]
  #pragma unroll
  for (int d = 0; d < 8; ++d) o[d] = acc[d] * r;
}

__global__ void k_r_talk_softmax(const float* __restrict__ Sr, const float* __restrict__ rt,
                                 float* __restrict__ Ar) {
  __shared__ float red[64];
  int rowid = blockIdx.x;               // (t,d,p) 4096
  int p = rowid & 63, d = (rowid >> 6) & 7, tt = rowid >> 9;
  int q = threadIdx.x;
  float acc = 0.f;
  for (int h = 0; h < 8; ++h)
    #pragma unroll
    for (int c = 0; c < 8; ++c) {
      float w = rt[(((size_t)h * 8 + c) * 8 + d) * 8 + tt];
      acc += w * Sr[(((size_t)h * 8 + c) * 64 + p) * 64 + q];
    }
  red[q] = acc;
  __syncthreads();
  float m = -1e30f;
  for (int i = 0; i < 64; ++i) m = fmaxf(m, red[i]);
  float e = __expf(acc - m);
  __syncthreads();
  red[q] = e;
  __syncthreads();
  float sum = 0.f;
  for (int i = 0; i < 64; ++i) sum += red[i];
  Ar[(size_t)rowid * 64 + q] = e / sum; // [t][d][p][q]
}

__global__ __launch_bounds__(256)
void k_out_lg(const float* __restrict__ Al, const float* __restrict__ Ag,
              const float* __restrict__ V, float* __restrict__ OutIm) {
  int idx = blockIdx.x * 256 + threadIdx.x;     // (h,n,p)
  int p = idx & 63, n = (idx >> 6) & 1023, h = idx >> 16;
  int ir = ((n >> 5) * 8 + (p >> 3) + 4) & 255;
  int ic = ((n & 31) * 8 + (p & 7) + 4) & 255;
  size_t pix = (size_t)ir * IMG + ic;
  size_t vbase = (size_t)idx * 32;
  const float* al = &Al[(size_t)(h * 1024 + n) * 64];   // [c][d]
  const float* ag = &Ag[(size_t)(h * 64 + p) * 64];     // [c][d]
  #pragma unroll
  for (int c = 0; c < 8; ++c) {
    float sl = 0.f, sg = 0.f;
    #pragma unroll
    for (int d = 0; d < 8; ++d) {
      sl += al[c * 8 + d] * V[vbase + 8 + d];
      sg += ag[c * 8 + d] * V[vbase + 16 + d];
    }
    OutIm[(size_t)(64 + h * 8 + c) * PIX + pix] = sl;
    OutIm[(size_t)(128 + h * 8 + c) * PIX + pix] = sg;
  }
}

__global__ __launch_bounds__(256)
void k_out_r(const float* __restrict__ Ar, const float* __restrict__ V,
             float* __restrict__ OutIm) {
  int idx = blockIdx.x * 256 + threadIdx.x;     // (h,n,p)
  int p = idx & 63, n = (idx >> 6) & 1023, h = idx >> 16;
  int ir = ((n >> 5) * 8 + (p >> 3) + 4) & 255;
  int ic = ((n & 31) * 8 + (p & 7) + 4) & 255;
  size_t pix = (size_t)ir * IMG + ic;
  size_t vbase = (size_t)(h * 1024 + n) * 2048;
  #pragma unroll
  for (int c = 0; c < 8; ++c) {
    const float* ar = &Ar[(((size_t)h * 8 + c) * 64 + p) * 64];
    float s = 0.f;
    for (int j = 0; j < 64; ++j)
      s += ar[j] * V[vbase + j * 32 + 24 + c];
    OutIm[(size_t)(192 + h * 8 + c) * PIX + pix] = s;
  }
}

// ---------------------------------------------------------------------------
// tail: gap -> sca, input scaling, dwu add, z gating
// ---------------------------------------------------------------------------
__global__ __launch_bounds__(256)
void k_gap(const float* __restrict__ OutIm, float* __restrict__ gap) {
  __shared__ float red[256];
  int ch = blockIdx.x, t = threadIdx.x;
  float s = 0.f;
  for (size_t p = t; p < PIX; p += 256) s += OutIm[(size_t)ch * PIX + p];
  red[t] = s;
  __syncthreads();
  for (int off = 128; off > 0; off >>= 1) {
    if (t < off) red[t] += red[t + off];
    __syncthreads();
  }
  if (t == 0) gap[ch] = red[0] * (1.0f / (float)PIX);
}

__global__ void k_sca(const float* __restrict__ gap, const float* __restrict__ wsca,
                      const float* __restrict__ bsca, float* __restrict__ sca) {
  int o = threadIdx.x;
  float s = bsca[o];
  for (int c = 0; c < 256; ++c) s += wsca[o * 256 + c] * gap[c];
  sca[o] = s;
}

__global__ __launch_bounds__(256)
void k_scale_rows(const float* __restrict__ OutIm, const float* __restrict__ sca,
                  float* __restrict__ Xs) {
  size_t i = (size_t)blockIdx.x * 256 + threadIdx.x;
  Xs[i] = OutIm[i] * sca[i >> 16];      // PIX == 1<<16
}

__global__ __launch_bounds__(256)
void k_add_dwu(float* __restrict__ Y1, const float* __restrict__ U,
               const float* __restrict__ wd) {
  size_t i = (size_t)blockIdx.x * 256 + threadIdx.x;
  int ch = (int)(i >> 16);
  int pix = (int)(i & 65535), r = pix >> 8, cc = pix & 255;
  float s = 0.f;
  #pragma unroll
  for (int di = -1; di <= 1; ++di) {
    int rr = r + di; if (rr < 0 || rr > 255) continue;
    #pragma unroll
    for (int dj = -1; dj <= 1; ++dj) {
      int c2 = cc + dj; if (c2 < 0 || c2 > 255) continue;
      s += wd[ch * 9 + (di + 1) * 3 + (dj + 1)] * U[((size_t)ch << 16) + rr * IMG + c2];
    }
  }
  Y1[i] += s;
}

__global__ __launch_bounds__(256)
void k_gate(const float* __restrict__ Y1, const float* __restrict__ Z,
            float* __restrict__ Xg) {
  size_t i = (size_t)blockIdx.x * 256 + threadIdx.x;
  Xg[i] = Y1[i] * Z[i];
}

// ---------------------------------------------------------------------------
extern "C" void kernel_launch(void* const* d_in, const int* in_sizes, int n_in,
                              void* d_out, int out_size, void* d_ws, size_t ws_size,
                              hipStream_t stream) {
  (void)in_sizes; (void)n_in; (void)out_size; (void)ws_size;
  const float* x          = (const float*)d_in[0];
  const float* temperature= (const float*)d_in[1];
  const float* r_talking  = (const float*)d_in[2];
  const float* g_talking  = (const float*)d_in[3];
  const float* b_talk_w   = (const float*)d_in[4];
  const float* l_talk_w   = (const float*)d_in[5];
  const float* w_quz      = (const float*)d_in[6];
  const float* w_kv       = (const float*)d_in[7];
  const float* w_dwu      = (const float*)d_in[8];
  const float* w_proj     = (const float*)d_in[9];
  const float* w_proj_out = (const float*)d_in[10];
  const float* w_sca      = (const float*)d_in[11];
  const float* b_sca      = (const float*)d_in[12];

  float* ws = (float*)d_ws;
  const size_t P = PIX;
  float* Yq   = ws;                         // 768P  (q_:0-255, u:256-511, z:512-767)
  float* Ykv  = Yq + 768 * P;               // 512P  (k_:0-255, v_:256-511)
  float* Q    = Ykv + 512 * P;              // 256P  (later reused as Xs)
  float* Kw   = Q + 256 * P;                // 256P  (later reused as Y1)
  float* V    = Kw + 256 * P;               // 256P  (later reused as Xg)
  float* nrm  = V + 256 * P;                // 16384
  float* Sb   = nrm + 16384;                // 512P
  float* Ab   = Sb + 512 * P;               // 512P
  float* Sl   = Ab + 512 * P;               // 8P
  float* Al   = Sl + 8 * P;                 // 8P
  float* Sg   = Al + 8 * P;                 // 32768
  float* Ag   = Sg + 32768;                 // 32768
  float* Sr   = Ag + 32768;                 // 4P
  float* Ar   = Sr + 4 * P;                 // 4P
  float* OutIm= Ar + 4 * P;                 // 256P
  float* gap  = OutIm + 256 * P;            // 256
  float* sca  = gap + 256;                  // 256

  const float* U = Yq + 256 * P;
  const float* Z = Yq + 512 * P;

  // 1) input projections (bf16 WMMA GEMM, TDM-staged B tiles)
  k_gemm_bf16<<<dim3(P / 128, 768 / 16), 256, 0, stream>>>(w_quz, x, Yq, 768, 128, (int)P);
  k_gemm_bf16<<<dim3(P / 128, 512 / 16), 256, 0, stream>>>(w_kv,  x, Ykv, 512, 128, (int)P);

  // 2) shift + window + normalize
  k_window_qkv<<<2048, 256, 0, stream>>>(Yq, Ykv, Q, Kw, V);
  k_norm_bl<<<8192, 64, 0, stream>>>(Q, Kw);
  k_norm_gr_reduce<<<512, 256, 0, stream>>>(Q, Kw, nrm);
  k_norm_gr_apply<<<32768, 256, 0, stream>>>(Q, Kw, nrm);

  // 3) attention scores
  k_attn_b_score<<<1024, 256, 0, stream>>>(Q, Kw, temperature, Sb);
  k_attn_l<<<8192, 64, 0, stream>>>(Q, Kw, temperature, Sl);
  k_attn_g<<<512, 64, 0, stream>>>(Q, Kw, temperature, Sg);
  k_attn_r<<<4096, 64, 0, stream>>>(Q, Kw, temperature, Sr);

  // 4) talking heads + mask + softmax
  k_talk_mask_softmax_b<<<131072, 256, 0, stream>>>(Sb, b_talk_w, Ab);
  k_talk_softmax_l<<<256, 256, 0, stream>>>(Sl, l_talk_w, Al);
  k_g_talk_softmax<<<16, 256, 0, stream>>>(Sg, g_talking, Ag);
  k_r_talk_softmax<<<4096, 64, 0, stream>>>(Sr, r_talking, Ar);

  // 5) attention outputs -> assembled (rolled) image
  k_out_b<<<1024, 256, 0, stream>>>(Ab, V, OutIm);
  k_out_lg<<<2048, 256, 0, stream>>>(Al, Ag, V, OutIm);
  k_out_r<<<2048, 256, 0, stream>>>(Ar, V, OutIm);

  // 6) sca gating + proj + dwu + z gate + final proj
  k_gap<<<256, 256, 0, stream>>>(OutIm, gap);
  k_sca<<<1, 256, 0, stream>>>(gap, w_sca, b_sca, sca);
  float* Xs = Q;   // reuse
  float* Y1 = Kw;  // reuse
  float* Xg = V;   // reuse
  k_scale_rows<<<65536, 256, 0, stream>>>(OutIm, sca, Xs);
  k_gemm_bf16<<<dim3(P / 128, 256 / 16), 256, 0, stream>>>(w_proj, Xs, Y1, 256, 256, (int)P);
  k_add_dwu<<<65536, 256, 0, stream>>>(Y1, U, w_dwu);
  k_gate<<<65536, 256, 0, stream>>>(Y1, Z, Xg);
  k_gemm_bf16<<<dim3(P / 128, 128 / 16), 256, 0, stream>>>(w_proj_out, Xg, (float*)d_out, 128, 256, (int)P);
}
